// SigCLossBase_88845693485550
// MI455X (gfx1250) — matrix operand order repaired and checked by reading, not compiled, not run on hardware
//
#include <hip/hip_runtime.h>

// SigLIP sigmoid-contrastive loss, N=16384, D=512, fused GEMM + softplus reduce.
// fp32 emulated on bf16 WMMA via hi/lo split (3 products) instead of
// native V_WMMA_F32_16X16X4_F32 at near-fp32 accuracy. Compute-bound
// (AI ~ 4 KFLOP/B); both 32MB feature matrices resident in 192MB L2.
// v2: register-prefetch double buffering (global latency hidden behind WMMA),
//     hardware-transcendental softplus (v_exp/v_log instead of log1pf libcall).

#define N_ROWS 16384
#define D_K    512
#define BM     128
#define BN     128
#define BK     64
#define LSTRIDE 72      // padded bf16 row stride in LDS (bank-conflict avoidance)
#define THREADS 256     // 8 wave32s
#define STAGE_ITERS ((BM * BK / 4) / THREADS)   // 8 float4 per matrix per thread

typedef __attribute__((ext_vector_type(16))) __bf16 bf16x16;
typedef __attribute__((ext_vector_type(8)))  __bf16 bf16x8;
typedef __attribute__((ext_vector_type(8)))  float  f32x8;
typedef __attribute__((ext_vector_type(4)))  unsigned short u16x4;

static __device__ __forceinline__ unsigned short bf16_rne(float x) {
  unsigned u = __float_as_uint(x);
  unsigned r = u + 0x7FFFu + ((u >> 16) & 1u);
  return (unsigned short)(r >> 16);
}

// x ~= bf16(hi) + bf16(lo); residual split for fp32-emulated bf16 GEMM
static __device__ __forceinline__ void split2(float x, unsigned short& hi,
                                              unsigned short& lo) {
  unsigned u = __float_as_uint(x);
  unsigned r = (u + 0x7FFFu + ((u >> 16) & 1u)) & 0xFFFF0000u;
  hi = (unsigned short)(r >> 16);
  float res = x - __uint_as_float(r);
  lo = bf16_rne(res);
}

// Build a v16bf WMMA fragment from two 16-byte LDS runs (two ds_load_b128)
static __device__ __forceinline__ bf16x16 frag2(const unsigned short* p0,
                                                const unsigned short* p1) {
  bf16x8 a = *(const bf16x8*)p0;
  bf16x8 b = *(const bf16x8*)p1;
  return __builtin_shufflevector(a, b, 0, 1, 2, 3, 4, 5, 6, 7,
                                 8, 9, 10, 11, 12, 13, 14, 15);
}

__global__ __launch_bounds__(THREADS)
void siglip_loss_main(const float* __restrict__ A,      // first_features  [N,D]
                      const float* __restrict__ B,      // second_features [N,D]
                      const int* __restrict__ lab1,     // first_label  [N]
                      const int* __restrict__ lab2,     // second_label [N]
                      const float* __restrict__ scale_p,
                      const float* __restrict__ bias_p,
                      float* __restrict__ partial) {
  __shared__ unsigned short AhiS[BM * LSTRIDE];
  __shared__ unsigned short AloS[BM * LSTRIDE];
  __shared__ unsigned short BhiS[BN * LSTRIDE];
  __shared__ unsigned short BloS[BN * LSTRIDE];
  __shared__ float red[THREADS];

  const int tid   = threadIdx.x;
  const int lane  = tid & 31;
  const int wave  = tid >> 5;
  const int waveM = wave >> 2;    // 0..1 : 64 output rows each
  const int waveN = wave & 3;     // 0..3 : 32 output cols each
  const int l16   = lane & 15;
  const int g     = lane >> 4;    // lane half (WMMA K-group select)

  const int bRow = blockIdx.y * BM;
  const int bCol = blockIdx.x * BN;

  f32x8 acc[4][2] = {};           // 4x2 tiles of 16x16, fp32 accumulation

  // Register double-buffer for the next K-chunk (hides global/L2 latency
  // behind the WMMA loop of the current chunk).
  float4 pa[STAGE_ITERS], pb[STAGE_ITERS];
  auto prefetch = [&](int k0) {
#pragma unroll
    for (int it = 0; it < STAGE_ITERS; ++it) {
      int q   = tid + it * THREADS;
      int row = q >> 4;           // 0..127
      int c4  = (q & 15) << 2;    // 0,4,...,60
      pa[it] = *(const float4*)(A + (size_t)(bRow + row) * D_K + k0 + c4);
      pb[it] = *(const float4*)(B + (size_t)(bCol + row) * D_K + k0 + c4);
    }
  };
  prefetch(0);

  for (int k0 = 0; k0 < D_K; k0 += BK) {
    // ---- commit prefetched chunk: fp32 -> bf16(hi,lo) split into LDS ----
#pragma unroll
    for (int it = 0; it < STAGE_ITERS; ++it) {
      int q   = tid + it * THREADS;
      int row = q >> 4;
      int c4  = (q & 15) << 2;
      unsigned short h0, h1, h2, h3, l0, l1, l2, l3;
      split2(pa[it].x, h0, l0); split2(pa[it].y, h1, l1);
      split2(pa[it].z, h2, l2); split2(pa[it].w, h3, l3);
      *(u16x4*)(&AhiS[row * LSTRIDE + c4]) = (u16x4){h0, h1, h2, h3};
      *(u16x4*)(&AloS[row * LSTRIDE + c4]) = (u16x4){l0, l1, l2, l3};
      split2(pb[it].x, h0, l0); split2(pb[it].y, h1, l1);
      split2(pb[it].z, h2, l2); split2(pb[it].w, h3, l3);
      *(u16x4*)(&BhiS[row * LSTRIDE + c4]) = (u16x4){h0, h1, h2, h3};
      *(u16x4*)(&BloS[row * LSTRIDE + c4]) = (u16x4){l0, l1, l2, l3};
    }
    __syncthreads();

    // issue next chunk's global loads now; WMMA loop below hides the latency
    if (k0 + BK < D_K) prefetch(k0 + BK);

    // ---- compute: 2 k-steps of 32, 24 WMMAs per k-step per wave ----
#pragma unroll
    for (int kk = 0; kk < BK; kk += 32) {
      bf16x16 ah[4], al[4], bh[2], bl[2];
      // A fragment (16x32, ISA layout): lane holds K runs [kk+8g, +8) and
      // [kk+16+8g, +8) of row (waveM*64 + tm*16 + l16)
#pragma unroll
      for (int tm = 0; tm < 4; ++tm) {
        int row = waveM * 64 + tm * 16 + l16;
        const unsigned short* p = &AhiS[row * LSTRIDE + kk + g * 8];
        ah[tm] = frag2(p, p + 16);
        const unsigned short* q = &AloS[row * LSTRIDE + kk + g * 8];
        al[tm] = frag2(q, q + 16);
      }
      // B fragment (32x16): lane holds K runs [kk+8g, +8) and [kk+16+8g, +8) of column
      // (waveN*32 + tn*16 + l16); column n of logits = row n of B
#pragma unroll
      for (int tn = 0; tn < 2; ++tn) {
        int col = waveN * 32 + tn * 16 + l16;
        const unsigned short* p = &BhiS[col * LSTRIDE + kk + g * 8];
        bh[tn] = frag2(p, p + 16);
        const unsigned short* q = &BloS[col * LSTRIDE + kk + g * 8];
        bl[tn] = frag2(q, q + 16);
      }
      // fp32-emulated product: hi*hi + hi*lo + lo*hi
#pragma unroll
      for (int tm = 0; tm < 4; ++tm)
#pragma unroll
        for (int tn = 0; tn < 2; ++tn) {
          acc[tm][tn] = __builtin_amdgcn_wmma_f32_16x16x32_bf16(
              false, ah[tm], false, bh[tn], (short)0, acc[tm][tn], false, false);
          acc[tm][tn] = __builtin_amdgcn_wmma_f32_16x16x32_bf16(
              false, ah[tm], false, bl[tn], (short)0, acc[tm][tn], false, false);
          acc[tm][tn] = __builtin_amdgcn_wmma_f32_16x16x32_bf16(
              false, al[tm], false, bh[tn], (short)0, acc[tm][tn], false, false);
        }
    }
    __syncthreads();   // compute done before next iteration overwrites LDS
  }

  // ---- fused epilogue: labels + stable softplus, per-block reduction ----
  // -logsigmoid(z) = softplus(-z) = max(x,0) + log(1 + exp(-|x|)), x = -z.
  // v_exp_f32/v_log_f32 hardware transcendentals; dropped log1p precision
  // only matters when exp(-|x|) < 2^-24 (|x|>16.6), an O(1e-7) abs error.
  const float scale = *scale_p;
  const float bias  = *bias_p;
  const int sl0 = lab2[bCol + waveN * 32 + l16];
  const int sl1 = lab2[bCol + waveN * 32 + 16 + l16];
  float lsum = 0.f;
#pragma unroll
  for (int tm = 0; tm < 4; ++tm) {
#pragma unroll
    for (int r = 0; r < 8; ++r) {
      // C layout: VGPR r, lane -> M = 8*(lane/16)+r, N = lane%16
      int ig = bRow + waveM * 64 + tm * 16 + g * 8 + r;
      int fl = lab1[ig];
#pragma unroll
      for (int tn = 0; tn < 2; ++tn) {
        int sl = (tn == 0) ? sl0 : sl1;
        float logit = fmaf(scale, acc[tm][tn][r], bias);
        float lab = (fl == sl) ? 1.0f : -1.0f;
        float x = -lab * logit;
        float e = __expf(-fabsf(x));
        lsum += fmaxf(x, 0.f) + __logf(1.0f + e);
      }
    }
  }
  red[tid] = lsum;
  __syncthreads();
#pragma unroll
  for (int off = THREADS / 2; off > 0; off >>= 1) {
    if (tid < off) red[tid] += red[tid + off];
    __syncthreads();
  }
  if (tid == 0) partial[blockIdx.y * gridDim.x + blockIdx.x] = red[0];
}

__global__ __launch_bounds__(THREADS)
void siglip_loss_reduce(const float* __restrict__ partial, float* __restrict__ out,
                        int n, float invN) {
  __shared__ float red[THREADS];
  float s = 0.f;
  for (int i = threadIdx.x; i < n; i += THREADS) s += partial[i];
  red[threadIdx.x] = s;
  __syncthreads();
#pragma unroll
  for (int off = THREADS / 2; off > 0; off >>= 1) {
    if (threadIdx.x < off) red[threadIdx.x] += red[threadIdx.x + off];
    __syncthreads();
  }
  if (threadIdx.x == 0) out[0] = red[0] * invN;
}

extern "C" void kernel_launch(void* const* d_in, const int* in_sizes, int n_in,
                              void* d_out, int out_size, void* d_ws, size_t ws_size,
                              hipStream_t stream) {
  (void)in_sizes; (void)n_in; (void)out_size; (void)ws_size;
  const float* A       = (const float*)d_in[0];
  const float* B       = (const float*)d_in[1];
  const int*   lab1    = (const int*)d_in[2];
  const int*   lab2    = (const int*)d_in[3];
  const float* scale_p = (const float*)d_in[4];
  const float* bias_p  = (const float*)d_in[5];
  float* out     = (float*)d_out;
  float* partial = (float*)d_ws;   // 16384 floats of block partials

  dim3 grid(N_ROWS / BN, N_ROWS / BM);   // 128 x 128 blocks
  siglip_loss_main<<<grid, THREADS, 0, stream>>>(A, B, lab1, lab2, scale_p,
                                                 bias_p, partial);
  siglip_loss_reduce<<<1, THREADS, 0, stream>>>(partial, out,
                                                (int)(grid.x * grid.y),
                                                1.0f / (float)N_ROWS);
}
